// SparseTransformer_83906481095480
// MI455X (gfx1250) — matrix operand order repaired and checked by reading, not compiled, not run on hardware
//
#include <hip/hip_runtime.h>
#include <hip/hip_bf16.h>
#include <math.h>

// ---------------- model config (matches reference) ----------------
#define LDEPTH 2
#define DIMM   512
#define HEADS  8
#define DHEAD  64
#define MLPD   2048
#define BATCH  4
#define SEQ    1024
#define NB     32
#define BLK    32
#define TOPKK  8
#define WRADI  32
#define MTOK   (BATCH * SEQ)   // 4096 tokens

typedef __attribute__((ext_vector_type(16))) __bf16 v16bf;
typedef __attribute__((ext_vector_type(8)))  __bf16 v8bf;
typedef __attribute__((ext_vector_type(8)))  float  v8f;

__device__ __forceinline__ v8f wmma_bf16(v16bf a, v16bf b, v8f c) {
  return __builtin_amdgcn_wmma_f32_16x16x32_bf16(false, a, false, b, (short)0, c,
                                                 false, false);
}
// two contiguous 8x bf16 (16B each) -> one 16-element fragment
__device__ __forceinline__ v16bf ld16(const __bf16* p0, const __bf16* p1) {
  const v8bf lo = *(const v8bf*)p0;
  const v8bf hi = *(const v8bf*)p1;
  return __builtin_shufflevector(lo, hi, 0, 1, 2, 3, 4, 5, 6, 7,
                                 8, 9, 10, 11, 12, 13, 14, 15);
}

// ---------------- fp32 [K,N] -> bf16 transposed [N,K] ----------------
__global__ void cvt_t_kernel(const float* __restrict__ src, __bf16* __restrict__ dst,
                             int K, int N) {
  int i = blockIdx.x * blockDim.x + threadIdx.x;
  if (i < K * N) {
    const int k = i / N, n = i - k * N;
    dst[(size_t)n * K + k] = (__bf16)src[i];
  }
}

// ---------------- V [tok][H*DH] bf16 -> Vt [b,h,dh,seq] bf16 ----------------
__global__ void vtrans_kernel(const __bf16* __restrict__ v, __bf16* __restrict__ vT) {
  const int i = blockIdx.x * blockDim.x + threadIdx.x;   // B*H*DH*SEQ = 2^21
  if (i >= BATCH * HEADS * DHEAD * SEQ) return;
  const int t = i & (SEQ - 1);
  const int d = (i >> 10) & (DHEAD - 1);
  const int h = (i >> 16) & (HEADS - 1);
  const int b = i >> 19;
  vT[i] = v[((size_t)(b * SEQ + t)) * DIMM + h * DHEAD + d];
}

// ---------------- LayerNorm (one block per token) ----------------
__global__ __launch_bounds__(256)
void ln_kernel(const float* __restrict__ x, const float* __restrict__ g,
               const float* __restrict__ bta, __bf16* __restrict__ out, int dim) {
  __shared__ float s1[256], s2[256];
  const int row = blockIdx.x;
  const float* xr = x + (size_t)row * dim;
  float a = 0.f, b = 0.f;
  for (int i = threadIdx.x; i < dim; i += blockDim.x) {
    float v = xr[i]; a += v; b += v * v;
  }
  s1[threadIdx.x] = a; s2[threadIdx.x] = b;
  __syncthreads();
  for (int st = 128; st > 0; st >>= 1) {
    if (threadIdx.x < st) {
      s1[threadIdx.x] += s1[threadIdx.x + st];
      s2[threadIdx.x] += s2[threadIdx.x + st];
    }
    __syncthreads();
  }
  const float mean = s1[0] / dim;
  const float var  = s2[0] / dim - mean * mean;
  const float inv  = rsqrtf(var + 1e-5f);
  for (int i = threadIdx.x; i < dim; i += blockDim.x)
    out[(size_t)row * dim + i] = (__bf16)((xr[i] - mean) * inv * g[i] + bta[i]);
}

// ------- bf16 WMMA GEMM: C = A[MxK] @ Wt[NxK]^T, 16x64 per wave ------------
// EPI 1: bf16 out, scaled (q/k/v). EPI 2: +bias, sigmoid, f32 (gates).
// EPI 3: +bias, tanh-GELU, bf16 (MLP up). EPI 4: +bias, +residual, f32.
template <int EPI>
__global__ __launch_bounds__(256)
void gemm_kernel(const __bf16* __restrict__ A, const __bf16* __restrict__ Wt,
                 const float* __restrict__ bias, const float* __restrict__ resid,
                 float* __restrict__ outF, __bf16* __restrict__ outH,
                 int Mm, int Nn, int Kk, float scl) {
  const int lane = threadIdx.x & 31;
  const int wave = blockIdx.x * (blockDim.x >> 5) + (threadIdx.x >> 5);
  const int nt64 = (Nn + 63) >> 6;
  const int mT = wave / nt64;
  const int nS = wave - mT * nt64;
  if (mT * 16 >= Mm) return;
  const int m0 = mT * 16, n0 = nS * 64;
  const int row  = m0 + (lane & 15);
  const int akb  = (lane & 16) ? 8 : 0;
  const int bkb  = (lane & 16) ? 8 : 0;
  const int colL = lane & 15;
  const int rowH = (lane & 16) ? 8 : 0;

  v8f acc[4] = {};
  for (int k0 = 0; k0 < Kk; k0 += 32) {
    const __bf16* ap = A + (size_t)row * Kk + k0 + akb;
    const v16bf a = ld16(ap, ap + 16);
#pragma unroll
    for (int j = 0; j < 4; ++j) {
      if (n0 + j * 16 >= Nn) continue;
      const int col = n0 + j * 16 + colL;
      v16bf b;
      if (col < Nn) {
        const __bf16* bp = Wt + (size_t)col * Kk + k0 + bkb;
        b = ld16(bp, bp + 16);
      } else {
#pragma unroll
        for (int i = 0; i < 16; ++i) b[i] = (__bf16)0.f;
      }
      acc[j] = wmma_bf16(a, b, acc[j]);
    }
  }
#pragma unroll
  for (int j = 0; j < 4; ++j) {
    const int col = n0 + j * 16 + colL;
    if (col >= Nn) continue;
#pragma unroll
    for (int i = 0; i < 8; ++i) {
      const int r = m0 + rowH + i;
      float v = acc[j][i];
      if (EPI >= 2) v += bias[col];
      if (EPI == 1) outH[(size_t)r * Nn + col] = (__bf16)(v * scl);
      if (EPI == 2) outF[(size_t)r * Nn + col] = 1.f / (1.f + __expf(-v));
      if (EPI == 3) {
        const float u = 0.7978845608f * (v + 0.044715f * v * v * v);
        outH[(size_t)r * Nn + col] = (__bf16)(0.5f * v * (1.f + tanhf(u)));
      }
      if (EPI == 4) outF[(size_t)r * Nn + col] = v + resid[(size_t)r * Nn + col];
    }
  }
}

// ---------------- block mean-pool of K/V (kc row-major, vc transposed) -------
__global__ __launch_bounds__(64)
void compress_kernel(const __bf16* __restrict__ kf, const __bf16* __restrict__ vf,
                     __bf16* __restrict__ kc, __bf16* __restrict__ vcT) {
  const int idx = blockIdx.x;             // b*H*NB
  const int m = idx % NB;
  const int h = (idx / NB) % HEADS;
  const int b = idx / (NB * HEADS);
  const int d = threadIdx.x;
  float sk = 0.f, sv = 0.f;
  for (int t = 0; t < BLK; ++t) {
    const size_t off = ((size_t)(b * SEQ + m * BLK + t)) * DIMM + h * DHEAD + d;
    sk += (float)kf[off]; sv += (float)vf[off];
  }
  const int bh = b * HEADS + h;
  kc[((size_t)bh * NB + m) * DHEAD + d]  = (__bf16)(sk * (1.f / BLK));
  vcT[((size_t)bh * DHEAD + d) * NB + m] = (__bf16)(sv * (1.f / BLK));
}

// ---------------- compressed branch + top-k selection (1 wave / 16 queries) --
__global__ __launch_bounds__(32)
void cattn_kernel(const __bf16* __restrict__ qf, const __bf16* __restrict__ kc,
                  const __bf16* __restrict__ vcT, float* __restrict__ outc,
                  unsigned* __restrict__ maskw) {
  const int QSTR = 88, ST = 34;
  __shared__ __bf16 qs[16 * QSTR];
  __shared__ float S[16 * ST];
  __shared__ float rcp[16];
  const int lane = threadIdx.x;
  const int qt = blockIdx.x;
  const int q0 = (qt & 63) * 16;
  const int bh = qt >> 6;
  const int h = bh & (HEADS - 1);
  const int b = bh / HEADS;
  const size_t tokbase = (size_t)b * SEQ;

  for (int i = lane; i < 16 * 64; i += 32) {
    const int r = i >> 6, c = i & 63;
    qs[r * QSTR + c] = qf[(tokbase + q0 + r) * (size_t)DIMM + h * DHEAD + c];
  }
  __syncthreads();

  const int row  = lane & 15;
  const int akb  = (lane & 16) ? 8 : 0;
  const int bkb  = (lane & 16) ? 8 : 0;
  const int colL = lane & 15;
  const int rowH = (lane & 16) ? 8 : 0;

  // S = q @ kc^T   (16 x 32)
#pragma unroll
  for (int j = 0; j < 2; ++j) {
    v8f acc = {};
#pragma unroll
    for (int kk = 0; kk < DHEAD; kk += 32) {
      const __bf16* qp = &qs[row * QSTR + kk + akb];
      const v16bf a = ld16(qp, qp + 16);
      const __bf16* bp = kc + ((size_t)bh * NB + j * 16 + colL) * DHEAD + kk + bkb;
      acc = wmma_bf16(a, ld16(bp, bp + 16), acc);
    }
#pragma unroll
    for (int i = 0; i < 8; ++i)
      S[(rowH + i) * ST + j * 16 + colL] = acc[i];
  }
  __syncthreads();

  if (lane < 16) {
    float mx = -3.4e38f;
    for (int m = 0; m < NB; ++m) mx = fmaxf(mx, S[lane * ST + m]);
    float sum = 0.f;
    for (int m = 0; m < NB; ++m) sum += __expf(S[lane * ST + m] - mx);
    unsigned msk = 0;
    for (int t = 0; t < TOPKK; ++t) {            // top-8 blocks
      float best = -3.4e38f; int bi = 0;
      for (int m = 0; m < NB; ++m) {
        if ((msk >> m) & 1) continue;
        const float s = S[lane * ST + m];
        if (s > best) { best = s; bi = m; }
      }
      msk |= 1u << bi;
    }
    maskw[(size_t)bh * SEQ + q0 + lane] = msk;
    rcp[lane] = 1.f / sum;
    for (int m = 0; m < NB; ++m) S[lane * ST + m] = __expf(S[lane * ST + m] - mx);
  }
  __syncthreads();

  // out_c = P @ vc   (16 x 64)
  v16bf a;
#pragma unroll
  for (int i = 0; i < 8; ++i) {
    a[i]     = (__bf16)S[row * ST + akb + i];
    a[8 + i] = (__bf16)S[row * ST + 16 + akb + i];
  }
#pragma unroll
  for (int j = 0; j < 4; ++j) {
    v8f acc = {};
    const __bf16* bp = vcT + ((size_t)bh * DHEAD + j * 16 + colL) * NB + bkb;
    acc = wmma_bf16(a, ld16(bp, bp + 16), acc);
#pragma unroll
    for (int i = 0; i < 8; ++i) {
      const int r = rowH + i;
      outc[(tokbase + q0 + r) * (size_t)DIMM + h * DHEAD + j * 16 + colL] =
          acc[i] * rcp[r];
    }
  }
}

// ------- fine attention: selected + window branches + gated combine ----------
__global__ __launch_bounds__(128)
void fattn_kernel(const __bf16* __restrict__ qf, const __bf16* __restrict__ kf,
                  const __bf16* __restrict__ vT, const float* __restrict__ outc,
                  const float* __restrict__ gatesf, const unsigned* __restrict__ maskw,
                  __bf16* __restrict__ attnb) {
  const int QSTR = 88, SSTR = 1036, PSTR = 1048;
  __shared__ __bf16 qs[16 * QSTR];
  __shared__ float  S[16 * SSTR];      // ~66 KB f32 score slab
  __shared__ __bf16 Ps[16 * PSTR];     // selected-branch probs (bf16)
  __shared__ __bf16 Pw[16 * PSTR];     // window-branch probs (bf16)
  __shared__ float pred[2][16][8];
  __shared__ float rmax[2][16];
  __shared__ float rscl[2][16];
  __shared__ unsigned smask[16];

  const int tid  = threadIdx.x;
  const int lane = tid & 31;
  const int w    = tid >> 5;
  const int qt = blockIdx.x;
  const int q0 = (qt & 63) * 16;
  const int bh = qt >> 6;
  const int h = bh & (HEADS - 1);
  const int b = bh / HEADS;
  const size_t tokbase = (size_t)b * SEQ;

  for (int i = tid; i < 16 * 64; i += 128) {
    const int r = i >> 6, c = i & 63;
    qs[r * QSTR + c] = qf[(tokbase + q0 + r) * (size_t)DIMM + h * DHEAD + c];
  }
  __syncthreads();

  const int row  = lane & 15;
  const int akb  = (lane & 16) ? 8 : 0;
  const int bkb  = (lane & 16) ? 8 : 0;
  const int colL = lane & 15;
  const int rowH = (lane & 16) ? 8 : 0;

  // ---- scores: wave w covers keys [w*256, w*256+256)
  for (int t = 0; t < 16; ++t) {
    const int n0 = w * 256 + t * 16;
    v8f acc = {};
#pragma unroll
    for (int kk = 0; kk < DHEAD; kk += 32) {
      const __bf16* qp = &qs[row * QSTR + kk + akb];
      const __bf16* bp = kf + (tokbase + n0 + colL) * (size_t)DIMM + h * DHEAD + kk + bkb;
      acc = wmma_bf16(ld16(qp, qp + 16), ld16(bp, bp + 16), acc);
    }
#pragma unroll
    for (int i = 0; i < 8; ++i)
      S[(rowH + i) * SSTR + n0 + colL] = acc[i];
  }
  __syncthreads();

  // ---- two-pass masked softmax stats (selected + window)
  const int qi   = tid >> 3;
  const int part = tid & 7;
  const int qg   = q0 + qi;
  const unsigned mb = maskw[(size_t)bh * SEQ + qg];
  float mS = -3.4e38f, mW = -3.4e38f;
  for (int k2 = part * 128; k2 < part * 128 + 128; ++k2) {
    const float s = S[qi * SSTR + k2];
    if ((mb >> (k2 >> 5)) & 1) mS = fmaxf(mS, s);
    const int d = (qg > k2) ? (qg - k2) : (k2 - qg);
    if (d <= WRADI) mW = fmaxf(mW, s);
  }
  pred[0][qi][part] = mS; pred[1][qi][part] = mW;
  __syncthreads();
  if (tid < 16) {
    float a0 = -3.4e38f, a1 = -3.4e38f;
    for (int p = 0; p < 8; ++p) {
      a0 = fmaxf(a0, pred[0][tid][p]); a1 = fmaxf(a1, pred[1][tid][p]);
    }
    rmax[0][tid] = a0; rmax[1][tid] = a1;
    smask[tid] = maskw[(size_t)bh * SEQ + q0 + tid];
  }
  __syncthreads();
  float sS = 0.f, sW = 0.f;
  const float Ms = rmax[0][qi], Mw = rmax[1][qi];
  for (int k2 = part * 128; k2 < part * 128 + 128; ++k2) {
    const float s = S[qi * SSTR + k2];
    if ((mb >> (k2 >> 5)) & 1) sS += __expf(s - Ms);
    const int d = (qg > k2) ? (qg - k2) : (k2 - qg);
    if (d <= WRADI) sW += __expf(s - Mw);
  }
  pred[0][qi][part] = sS; pred[1][qi][part] = sW;
  __syncthreads();
  if (tid < 16) {
    float a0 = 0.f, a1 = 0.f;
    for (int p = 0; p < 8; ++p) { a0 += pred[0][tid][p]; a1 += pred[1][tid][p]; }
    rscl[0][tid] = 1.f / a0; rscl[1][tid] = 1.f / a1;
  }
  __syncthreads();

  // ---- materialize both P matrices once (bf16), shared by all waves
  for (int i = tid; i < 16 * SEQ; i += 128) {
    const int r = i >> 10, key = i & (SEQ - 1);
    const float s = S[r * SSTR + key];
    const bool sel = (smask[r] >> (key >> 5)) & 1;
    const int qr = q0 + r;
    const int d = (qr > key) ? (qr - key) : (key - qr);
    Ps[r * PSTR + key] = (__bf16)(sel ? __expf(s - rmax[0][r]) : 0.f);
    Pw[r * PSTR + key] = (__bf16)((d <= WRADI) ? __expf(s - rmax[1][r]) : 0.f);
  }
  __syncthreads();

  // ---- P @ V for both branches; wave w owns dh columns [w*16, w*16+16)
  v8f aS = {}, aW = {};
  const __bf16* vrow = vT + ((size_t)bh * DHEAD + w * 16 + colL) * SEQ;
  for (int ks = 0; ks < 32; ++ks) {
    const __bf16* ps = &Ps[row * PSTR + ks * 32 + akb];
    const __bf16* pw = &Pw[row * PSTR + ks * 32 + akb];
    const __bf16* bp = vrow + ks * 32 + bkb;
    const v16bf bfr = ld16(bp, bp + 16);
    aS = wmma_bf16(ld16(ps, ps + 16), bfr, aS);
    aW = wmma_bf16(ld16(pw, pw + 16), bfr, aW);
  }

  // ---- gated combine with compressed branch, bf16 out for Wo GEMM
#pragma unroll
  for (int i = 0; i < 8; ++i) {
    const int r = rowH + i;
    const size_t tok = tokbase + q0 + r;
    const float g0 = gatesf[tok * (HEADS * 3) + h * 3 + 0];
    const float g1 = gatesf[tok * (HEADS * 3) + h * 3 + 1];
    const float g2 = gatesf[tok * (HEADS * 3) + h * 3 + 2];
    const int cd = w * 16 + colL;
    const float oc = outc[tok * (size_t)DIMM + h * DHEAD + cd];
    const float val = g0 * oc + g1 * aS[i] * rscl[0][r] + g2 * aW[i] * rscl[1][r];
    attnb[tok * (size_t)DIMM + h * DHEAD + cd] = (__bf16)val;
  }
}

// ---------------- host orchestration ----------------
extern "C" void kernel_launch(void* const* d_in, const int* in_sizes, int n_in,
                              void* d_out, int out_size, void* d_ws, size_t ws_size,
                              hipStream_t stream) {
  const float* x    = (const float*)d_in[0];
  const float* ln1g = (const float*)d_in[1];
  const float* ln1b = (const float*)d_in[2];
  const float* Wq   = (const float*)d_in[3];
  const float* Wk   = (const float*)d_in[4];
  const float* Wv   = (const float*)d_in[5];
  const float* Wg   = (const float*)d_in[6];
  const float* bg   = (const float*)d_in[7];
  const float* Wo   = (const float*)d_in[8];
  const float* bo   = (const float*)d_in[9];
  const float* ln2g = (const float*)d_in[10];
  const float* ln2b = (const float*)d_in[11];
  const float* W1   = (const float*)d_in[12];
  const float* b1   = (const float*)d_in[13];
  const float* W2   = (const float*)d_in[14];
  const float* b2   = (const float*)d_in[15];
  float* xbuf = (float*)d_out;

  char* base = (char*)d_ws;
  size_t off = 0;
  auto alloc = [&](size_t bytes) -> void* {
    off = (off + 255) & ~(size_t)255;
    void* p = base + off;
    off += bytes;
    return p;
  };
  const size_t SQ = (size_t)DIMM * DIMM;
  __bf16* wqT = (__bf16*)alloc(sizeof(__bf16) * LDEPTH * SQ);
  __bf16* wkT = (__bf16*)alloc(sizeof(__bf16) * LDEPTH * SQ);
  __bf16* wvT = (__bf16*)alloc(sizeof(__bf16) * LDEPTH * SQ);
  __bf16* wgT = (__bf16*)alloc(sizeof(__bf16) * LDEPTH * DIMM * HEADS * 3);
  __bf16* woT = (__bf16*)alloc(sizeof(__bf16) * LDEPTH * SQ);
  __bf16* w1T = (__bf16*)alloc(sizeof(__bf16) * (size_t)LDEPTH * DIMM * MLPD);
  __bf16* w2T = (__bf16*)alloc(sizeof(__bf16) * (size_t)LDEPTH * MLPD * DIMM);
  __bf16* hln = (__bf16*)alloc(sizeof(__bf16) * (size_t)MTOK * DIMM);
  __bf16* qb  = (__bf16*)alloc(sizeof(__bf16) * (size_t)MTOK * DIMM);
  __bf16* kb  = (__bf16*)alloc(sizeof(__bf16) * (size_t)MTOK * DIMM);
  __bf16* vb  = (__bf16*)alloc(sizeof(__bf16) * (size_t)MTOK * DIMM);
  __bf16* vT  = (__bf16*)alloc(sizeof(__bf16) * (size_t)MTOK * DIMM);
  float* gatesf = (float*)alloc(sizeof(float) * (size_t)MTOK * HEADS * 3);
  __bf16* kc  = (__bf16*)alloc(sizeof(__bf16) * (size_t)BATCH * HEADS * NB * DHEAD);
  __bf16* vcT = (__bf16*)alloc(sizeof(__bf16) * (size_t)BATCH * HEADS * DHEAD * NB);
  unsigned* maskw = (unsigned*)alloc(sizeof(unsigned) * (size_t)BATCH * HEADS * SEQ);
  float* outc   = (float*)alloc(sizeof(float) * (size_t)MTOK * DIMM);
  __bf16* attnb = (__bf16*)alloc(sizeof(__bf16) * (size_t)MTOK * DIMM);
  __bf16* midb  = (__bf16*)alloc(sizeof(__bf16) * (size_t)MTOK * MLPD);

  auto cvtT = [&](const float* s, __bf16* d, int K, int N) {
    const int tot = K * N;
    cvt_t_kernel<<<(unsigned)((tot + 255) / 256), 256, 0, stream>>>(s, d, K, N);
  };
  for (int l = 0; l < LDEPTH; ++l) {
    cvtT(Wq + l * SQ, wqT + l * SQ, DIMM, DIMM);
    cvtT(Wk + l * SQ, wkT + l * SQ, DIMM, DIMM);
    cvtT(Wv + l * SQ, wvT + l * SQ, DIMM, DIMM);
    cvtT(Wg + (size_t)l * DIMM * HEADS * 3, wgT + (size_t)l * DIMM * HEADS * 3,
         DIMM, HEADS * 3);
    cvtT(Wo + l * SQ, woT + l * SQ, DIMM, DIMM);
    cvtT(W1 + (size_t)l * DIMM * MLPD, w1T + (size_t)l * DIMM * MLPD, DIMM, MLPD);
    cvtT(W2 + (size_t)l * MLPD * DIMM, w2T + (size_t)l * MLPD * DIMM, MLPD, DIMM);
  }

  hipMemcpyAsync(xbuf, x, sizeof(float) * (size_t)MTOK * DIMM,
                 hipMemcpyDeviceToDevice, stream);

  auto gemm = [&](int epi, const __bf16* A, const __bf16* Wt, const float* bias,
                  const float* resid, float* oF, __bf16* oH, int Mm, int Nn, int Kk,
                  float scl) {
    const int nt64 = (Nn + 63) / 64;
    const unsigned blocks = (unsigned)(((Mm / 16) * nt64 + 7) / 8);
    switch (epi) {
      case 1: gemm_kernel<1><<<blocks, 256, 0, stream>>>(A, Wt, bias, resid, oF, oH, Mm, Nn, Kk, scl); break;
      case 2: gemm_kernel<2><<<blocks, 256, 0, stream>>>(A, Wt, bias, resid, oF, oH, Mm, Nn, Kk, scl); break;
      case 3: gemm_kernel<3><<<blocks, 256, 0, stream>>>(A, Wt, bias, resid, oF, oH, Mm, Nn, Kk, scl); break;
      case 4: gemm_kernel<4><<<blocks, 256, 0, stream>>>(A, Wt, bias, resid, oF, oH, Mm, Nn, Kk, scl); break;
    }
  };

  const float qscale = 0.125f;   // DH^-0.5
  for (int l = 0; l < LDEPTH; ++l) {
    const size_t wofs = l * SQ;
    ln_kernel<<<MTOK, 256, 0, stream>>>(xbuf, ln1g + l * DIMM, ln1b + l * DIMM, hln, DIMM);
    gemm(1, hln, wqT + wofs, nullptr, nullptr, nullptr, qb, MTOK, DIMM, DIMM, qscale);
    gemm(1, hln, wkT + wofs, nullptr, nullptr, nullptr, kb, MTOK, DIMM, DIMM, 1.f);
    gemm(1, hln, wvT + wofs, nullptr, nullptr, nullptr, vb, MTOK, DIMM, DIMM, 1.f);
    gemm(2, hln, wgT + (size_t)l * DIMM * HEADS * 3, bg + l * HEADS * 3, nullptr,
         gatesf, nullptr, MTOK, HEADS * 3, DIMM, 1.f);
    vtrans_kernel<<<(MTOK * DIMM) / 256, 256, 0, stream>>>(vb, vT);
    compress_kernel<<<BATCH * HEADS * NB, 64, 0, stream>>>(kb, vb, kc, vcT);
    cattn_kernel<<<BATCH * HEADS * (SEQ / 16), 32, 0, stream>>>(qb, kc, vcT, outc, maskw);
    fattn_kernel<<<BATCH * HEADS * (SEQ / 16), 128, 0, stream>>>(qb, kb, vT, outc,
                                                                 gatesf, maskw, attnb);
    gemm(4, attnb, woT + wofs, bo + l * DIMM, xbuf, xbuf, nullptr, MTOK, DIMM, DIMM, 1.f);
    ln_kernel<<<MTOK, 256, 0, stream>>>(xbuf, ln2g + l * DIMM, ln2b + l * DIMM, hln, DIMM);
    gemm(3, hln, w1T + (size_t)l * DIMM * MLPD, b1 + l * MLPD, nullptr, nullptr,
         midb, MTOK, MLPD, DIMM, 1.f);
    gemm(4, midb, w2T + (size_t)l * MLPD * DIMM, b2 + l * DIMM, xbuf, xbuf, nullptr,
         MTOK, DIMM, MLPD, 1.f);
  }
}
